// SelfAttention_8040178778426
// MI455X (gfx1250) — matrix operand order repaired and checked by reading, not compiled, not run on hardware
//
#include <hip/hip_runtime.h>
#include <hip/hip_bf16.h>

// ---------------------------------------------------------------------------
// 2-layer transformer encoder forward for MI455X (gfx1250, wave32, WMMA).
// - All matmuls on v_wmma_f32_16x16x32_f16 (f16 in, f32 accum).
// - Weights pre-swizzled in global into WMMA B-fragment order.
// - GEMM: A tile async-to-LDS (ASYNCcnt), B panel via Tensor Data Mover
//   (TENSORcnt) into LDS; both double-buffered. Branchless templated epilogue.
// - V-projection GEMM writes fragment-order output so flash attention loads
//   V fragments directly from global (no LDS scatter).
// ---------------------------------------------------------------------------

typedef __attribute__((ext_vector_type(16))) _Float16 v16h;
typedef __attribute__((ext_vector_type(8)))  _Float16 v8h;
typedef __attribute__((ext_vector_type(8)))  float    v8f;
typedef __attribute__((ext_vector_type(4)))  unsigned int u32x4;
typedef __attribute__((ext_vector_type(4)))  int          i32x4;
typedef __attribute__((ext_vector_type(8)))  int          i32x8;

#ifndef __has_builtin
#define __has_builtin(x) 0
#endif
#if __has_builtin(__builtin_amdgcn_tensor_load_to_lds)
#define HAVE_TDM 1
#else
#define HAVE_TDM 0
#endif

#define B_      4
#define L_      1024
#define D_      512
#define H_      8
#define DH_     64
#define DI_     1024
#define NROWS   (B_ * L_)          // 4096

// GEMM epilogue modes
#define OUT_F32      0
#define OUT_F16ROW   1
#define OUT_F16FRAG  2

__device__ __forceinline__ v16h combine16(v8h lo, v8h hi) {
    return __builtin_shufflevector(lo, hi, 0,1,2,3,4,5,6,7,8,9,10,11,12,13,14,15);
}
__device__ __forceinline__ v16h ld32B(const _Float16* p) {     // 32 contiguous bytes
    return combine16(*(const v8h*)p, *(const v8h*)(p + 8));
}
__device__ __forceinline__ v8f wmma_f16(v16h a, v16h b, v8f c) {
    return __builtin_amdgcn_wmma_f32_16x16x32_f16(false, a, false, b,
                                                  (short)0, c, false, false);
}

// CDNA5 async copy: global -> LDS, 16 bytes per lane, tracked by ASYNCcnt.
__device__ __forceinline__ void async_ld_b128(unsigned ldsOff, const _Float16* g) {
    asm volatile("global_load_async_to_lds_b128 %0, %1, off"
                 :: "v"(ldsOff), "v"(g) : "memory");
}
__device__ __forceinline__ void wait_async0() {
    asm volatile("s_wait_asynccnt 0x0" ::: "memory");
}

#if HAVE_TDM
// Tensor Data Mover: 1-D contiguous copy of `nhalves` f16 elements -> LDS.
// D# per ISA ch.8: group0 {count=1, lds_addr, global_addr, type=2},
// group1 {data_size=2B, tensor dims large, tile_dim0=nhalves}.
__device__ __forceinline__ void tdm_load_1d(unsigned ldsOff, const void* gaddr,
                                            unsigned nhalves) {
    unsigned long long ga = (unsigned long long)(uintptr_t)gaddr;
    u32x4 g0 = {};
    g0[0] = 1u;                                   // count=1 (valid descriptor)
    g0[1] = ldsOff;                               // LDS byte address
    g0[2] = (unsigned)ga;                         // global_addr[31:0]
    g0[3] = (unsigned)(ga >> 32) | 0x80000000u;   // global_addr[56:32] | type=2
    i32x8 g1 = {};
    g1[0] = (1 << 16);                            // data_size = 2 bytes
    g1[1] = 0;                                    // tensor_dim0 low16 (=0)
    g1[2] = 0x0100;                               // tensor_dim0 = 1<<24 (no OOB)
    g1[3] = 0x10 | (int)(nhalves << 16);          // tensor_dim1=1<<20, tile_dim0
    g1[4] = 0;                                    // tile_dim1/2 unused (1-D)
    g1[5] = 0x01000000;                           // tensor_dim0_stride (don't care)
    i32x4 g2 = {}, g3 = {};
#if defined(__clang_major__) && (__clang_major__ >= 23)
    i32x8 g4 = {};
    __builtin_amdgcn_tensor_load_to_lds(g0, g1, g2, g3, g4, 0);
#else
    __builtin_amdgcn_tensor_load_to_lds(g0, g1, g2, g3, 0);
#endif
}
#endif

// WMMA B-fragment address for element (k, n) of a [K x N] operand, N mult of 16.
// Tile (k>>5, n>>4) is 512 contiguous halves in [lane][16] order:
//   lane = (n&15) + 16*((k>>4)&1), half-index = k&15.
__device__ __forceinline__ size_t bfrag_idx(int k, int n, int N) {
    return ((size_t)((k >> 5) * (N >> 4) + (n >> 4)) << 9)
         + ((size_t)(((n & 15) + (((k >> 4) & 1) << 4))) << 4)
         + (size_t)(k & 15);
}

// ---------------------------------------------------------------------------
// x = x + pos_enc  (fp32 out + f16 mirror)
// ---------------------------------------------------------------------------
__global__ void add_pos_kernel(const float* __restrict__ x,
                               const float* __restrict__ pe,
                               float* __restrict__ xf,
                               _Float16* __restrict__ xh) {
    size_t i  = (size_t)blockIdx.x * 256 + threadIdx.x;   // < B*L*D
    size_t ld = i % ((size_t)L_ * D_);
    float v = x[i] + pe[ld];
    xf[i] = v;
    xh[i] = (_Float16)v;
}

// ---------------------------------------------------------------------------
// w_qs[h][d][j] (fp32) -> f16 B-fragment order, B[k=d][n=h*64+j], N=512
// ---------------------------------------------------------------------------
__global__ void repack_qkv_frag_kernel(const float* __restrict__ src,
                                       _Float16* __restrict__ dst) {
    int i = blockIdx.x * 256 + threadIdx.x;               // < 8*512*64
    int j = i & 63;
    int d = (i >> 6) & 511;
    int h = i >> 15;
    dst[bfrag_idx(d, h * DH_ + j, D_)] = (_Float16)src[i];
}

// src[R][C] fp32; GEMM wants B[k][n] = src[n][k]  (n=r in 0..R-1, k=c, N=R)
__global__ void repack_T_frag_kernel(const float* __restrict__ src,
                                     _Float16* __restrict__ dst, int R, int C) {
    int i = blockIdx.x * 256 + threadIdx.x;               // < R*C
    int r = i / C, c = i - r * C;
    dst[bfrag_idx(c, r, R)] = (_Float16)src[i];
}

// ---------------------------------------------------------------------------
// WMMA GEMM: C[M x N] = A[M x K](f16 row-major) * B(f16, fragment order)
// Block tile 128x128, BK=32, 8 waves of 32x64 (2x4 WMMA tiles each).
// A tile: per-lane async-to-LDS. B panel (8KB contiguous): TDM to LDS.
// Both double-buffered. MODE/HASBIAS/RELU are compile-time (branchless tail).
// ---------------------------------------------------------------------------
template <int MODE, bool HASBIAS, bool RELU>
__global__ __launch_bounds__(256)
void gemm_wmma_kernel(const _Float16* __restrict__ A,
                      const _Float16* __restrict__ Bf,
                      void* __restrict__ outP,
                      const float* __restrict__ bias,
                      int M, int N, int K) {
    __shared__ _Float16 ldsA[2][128 * 40];   // row stride 40 halves (16B aligned)
    __shared__ _Float16 ldsB[2][4096];       // 8 fragment tiles per buffer

    const int tid  = threadIdx.x;
    const int lane = tid & 31;
    const int wid  = tid >> 5;
    const int wr   = wid >> 1;               // 0..3 : 32-row quadrant
    const int wc   = wid & 1;                // 0..1 : 64-col half
    const int m0   = blockIdx.y * 128;
    const int n0   = blockIdx.x * 128;

    const int arow = tid >> 1;               // 0..127
    const int aseg = (tid & 1) << 4;         // 0 or 16 halves

    const _Float16* gaBase = A + (size_t)(m0 + arow) * K + aseg;
    const unsigned ldsAoff[2] = {
        (unsigned)(uintptr_t)&ldsA[0][arow * 40 + aseg],
        (unsigned)(uintptr_t)&ldsA[1][arow * 40 + aseg] };
    const unsigned ldsBoff[2] = {
        (unsigned)(uintptr_t)&ldsB[0][0],
        (unsigned)(uintptr_t)&ldsB[1][0] };

    v8f acc[2][4] = {};

    auto stage_A = [&](int bsel, int kk) {
        const _Float16* g = gaBase + kk;
        async_ld_b128(ldsAoff[bsel],      g);
        async_ld_b128(ldsAoff[bsel] + 16, g + 8);
    };
    auto stage_B = [&](int bsel, int kk) {
        const _Float16* g = Bf + (((size_t)(kk >> 5) * (N >> 4) + (n0 >> 4)) << 9);
#if HAVE_TDM
        if (wid == 0) tdm_load_1d(ldsBoff[bsel], g, 4096);
#else
        unsigned o = ldsBoff[bsel] + (unsigned)tid * 32;
        async_ld_b128(o,      g + tid * 16);
        async_ld_b128(o + 16, g + tid * 16 + 8);
#endif
    };
    auto wait_stage = [&]() {
        wait_async0();
#if HAVE_TDM
        if (wid == 0) __builtin_amdgcn_s_wait_tensorcnt(0);
#endif
    };

    // prologue: stage k-tile 0 into buffer 0
    stage_A(0, 0);
    stage_B(0, 0);
    wait_stage();
    __syncthreads();

    int buf = 0;
    for (int k0 = 0; k0 < K; k0 += 32) {
        if (k0 + 32 < K) {                   // prefetch next k-tile
            stage_A(buf ^ 1, k0 + 32);
            stage_B(buf ^ 1, k0 + 32);
        }

        const int hs = (lane >> 4) << 4;     // 0 or 16 halves
        #pragma unroll
        for (int si = 0; si < 2; ++si) {
            int row = wr * 32 + si * 16 + (lane & 15);
            v8h alo = *(const v8h*)&ldsA[buf][row * 40 + hs];
            v8h ahi = *(const v8h*)&ldsA[buf][row * 40 + hs + 8];
            v16h af = combine16(alo, ahi);
            #pragma unroll
            for (int sj = 0; sj < 4; ++sj) {
                int nsub = wc * 4 + sj;      // 0..7
                v16h bfr = ld32B(&ldsB[buf][(nsub << 9) + (lane << 4)]);
                acc[si][sj] = wmma_f16(af, bfr, acc[si][sj]);
            }
        }

        wait_stage();
        __syncthreads();
        buf ^= 1;
    }

    // epilogue (branchless): C layout m = r + 8*(lane>=16), n = lane&15
    const int ncol = lane & 15;
    const int mh   = (lane >> 4) << 3;
    #pragma unroll
    for (int si = 0; si < 2; ++si) {
        #pragma unroll
        for (int sj = 0; sj < 4; ++sj) {
            const int gcol = n0 + wc * 64 + sj * 16 + ncol;
            const float bvv = HASBIAS ? bias[gcol] : 0.0f;
            #pragma unroll
            for (int r = 0; r < 8; ++r) {
                const int grow = m0 + wr * 32 + si * 16 + r + mh;
                float v = acc[si][sj][r] + bvv;
                if (RELU) v = v > 0.0f ? v : 0.0f;
                if (MODE == OUT_F32) {
                    ((float*)outP)[(size_t)grow * N + gcol] = v;
                } else if (MODE == OUT_F16ROW) {
                    ((_Float16*)outP)[(size_t)grow * N + gcol] = (_Float16)v;
                } else {
                    ((_Float16*)outP)[bfrag_idx(grow, gcol, N)] = (_Float16)v;
                }
            }
        }
    }
}

// ---------------------------------------------------------------------------
// Flash attention: one wave per (head, batch, 16-query tile).
// q/k: f16 [B*L, 512] row-major. v: f16 in B-fragment order over [B*L x 512].
// out: fp32 [B, L, 512] (heads concatenated).
// ---------------------------------------------------------------------------
__global__ __launch_bounds__(32)
void flash_attn_kernel(const _Float16* __restrict__ qh,
                       const _Float16* __restrict__ kh,
                       const _Float16* __restrict__ vfrag,
                       float* __restrict__ out) {
    __shared__ _Float16 ldsP[32 * 16];   // P in A-fragment order

    const int lane = threadIdx.x;
    int blk = blockIdx.x;                // ((h*B + b)*(L/16) + qt)
    const int qt = blk & 63;  blk >>= 6;
    const int b  = blk & 3;   blk >>= 2;
    const int h  = blk;
    const int q0 = qt * 16;
    const int m    = lane & 15;
    const int half = lane >> 4;
    const float scale = 0.04419417382415922f;    // 1/sqrt(D_MODEL=512)

    // Q fragments (16x32 A-layout), loaded once
    const _Float16* qrow = qh + ((size_t)(b * L_) + q0 + m) * D_ + h * DH_;
    v16h qf[2];
    #pragma unroll
    for (int ks = 0; ks < 2; ++ks) {
        v8h lo = *(const v8h*)(qrow + ks * 32 + half * 16);
        v8h hi = *(const v8h*)(qrow + ks * 32 + half * 16 + 8);
        qf[ks] = combine16(lo, hi);
    }

    v8f o[4] = {};
    float mrow[8], lrow[8];
    #pragma unroll
    for (int r = 0; r < 8; ++r) { mrow[r] = -1e30f; lrow[r] = 0.0f; }

    for (int j = 0; j < L_; j += 32) {
        // ---- S = Q K^T (two 16x16 tiles, each K-reduced over d_head=64) ----
        v8f s[2];
        #pragma unroll
        for (int t = 0; t < 2; ++t) {
            const _Float16* krow =
                kh + ((size_t)(b * L_) + j + t * 16 + m) * D_ + h * DH_;
            v8f c = {};
            #pragma unroll
            for (int ks = 0; ks < 2; ++ks) {
                v16h kf = ld32B(krow + ks * 32 + half * 16);
                c = wmma_f16(qf[ks], kf, c);
            }
            s[t] = c;
        }

        // ---- online softmax over rows (row = r + 8*half) ----
        float pbuf[2][8], corr[8];
        #pragma unroll
        for (int r = 0; r < 8; ++r) {
            float v0 = s[0][r] * scale, v1 = s[1][r] * scale;
            float mx = fmaxf(v0, v1);
            #pragma unroll
            for (int d2 = 1; d2 < 16; d2 <<= 1)
                mx = fmaxf(mx, __shfl_xor(mx, d2, 32));
            float mnew = fmaxf(mrow[r], mx);
            float c  = __expf(mrow[r] - mnew);
            float p0 = __expf(v0 - mnew);
            float p1 = __expf(v1 - mnew);
            float ps = p0 + p1;
            #pragma unroll
            for (int d2 = 1; d2 < 16; d2 <<= 1)
                ps += __shfl_xor(ps, d2, 32);
            lrow[r] = lrow[r] * c + ps;
            mrow[r] = mnew;
            corr[r] = c;
            pbuf[0][r] = p0; pbuf[1][r] = p1;
        }

        __syncthreads();
        // ---- scatter P (C layout -> A-fragment layout) ----
        #pragma unroll
        for (int t = 0; t < 2; ++t) {
            #pragma unroll
            for (int r = 0; r < 8; ++r) {
                int row = r + (half << 3);
                int key = t * 16 + m;
                int ht  = (key >> 4) & 1;
                int idx = key & 15;
                ldsP[((row + (ht << 4)) << 4) + idx] = (_Float16)pbuf[t][r];
            }
        }
        __syncthreads();

        // ---- O = O*corr + P V ; V fragments direct from global ----
        v16h pf = ld32B(&ldsP[lane << 4]);
        const _Float16* vtile =
            vfrag + ((size_t)(((b * L_ + j) >> 5) * (D_ >> 4) + h * 4) << 9);
        #pragma unroll
        for (int t = 0; t < 4; ++t) {
            #pragma unroll
            for (int r = 0; r < 8; ++r) o[t][r] *= corr[r];
            v16h vf = ld32B(vtile + ((size_t)t << 9) + (lane << 4));
            o[t] = wmma_f16(pf, vf, o[t]);
        }
    }

    // ---- normalize and write (concat heads at col h*64) ----
    float* orow = out + ((size_t)(b * L_) + q0) * D_ + h * DH_;
    #pragma unroll
    for (int t = 0; t < 4; ++t) {
        #pragma unroll
        for (int r = 0; r < 8; ++r) {
            int row = r + (half << 3);
            orow[(size_t)row * D_ + t * 16 + m] = o[t][r] / lrow[r];
        }
    }
}

// ---------------------------------------------------------------------------
// LayerNorm (torch-style: unbiased std, eps on sigma): one wave per row.
// y = (z - mu)/(sigma + 1e-3) * a + b  with z = z1 + res.
// WRITE16 compile-time to keep the tail branchless.
// ---------------------------------------------------------------------------
template <bool WRITE16>
__global__ __launch_bounds__(256)
void layernorm_kernel(const float* z1, const float* res,
                      const float* __restrict__ ga, const float* __restrict__ be,
                      float* outF32, _Float16* outF16) {
    const int lane = threadIdx.x & 31;
    const int wid  = threadIdx.x >> 5;
    const size_t row = (size_t)blockIdx.x * 8 + wid;     // < 4096
    const float* p1 = z1  + row * D_;
    const float* p2 = res + row * D_;

    float z[16], s = 0.0f;
    #pragma unroll
    for (int i = 0; i < 16; ++i) {
        int d = lane + i * 32;
        z[i] = p1[d] + p2[d];
        s += z[i];
    }
    #pragma unroll
    for (int d2 = 1; d2 < 32; d2 <<= 1) s += __shfl_xor(s, d2, 32);
    float mu = s * (1.0f / 512.0f);

    float ss = 0.0f;
    #pragma unroll
    for (int i = 0; i < 16; ++i) { float t = z[i] - mu; ss += t * t; }
    #pragma unroll
    for (int d2 = 1; d2 < 32; d2 <<= 1) ss += __shfl_xor(ss, d2, 32);
    float sigma = sqrtf(ss * (1.0f / 511.0f));
    float inv = 1.0f / (sigma + 1e-3f);

    #pragma unroll
    for (int i = 0; i < 16; ++i) {
        int d = lane + i * 32;
        float y = (z[i] - mu) * inv * ga[d] + be[d];
        outF32[row * D_ + d] = y;
        if (WRITE16) outF16[row * D_ + d] = (_Float16)y;
    }
}

// ---------------------------------------------------------------------------
// Host orchestration
// ---------------------------------------------------------------------------
extern "C" void kernel_launch(void* const* d_in, const int* in_sizes, int n_in,
                              void* d_out, int out_size, void* d_ws, size_t ws_size,
                              hipStream_t stream) {
    (void)in_sizes; (void)n_in; (void)out_size; (void)ws_size;
    const float* x     = (const float*)d_in[0];
    const float* pe    = (const float*)d_in[1];
    const float* w_qs  = (const float*)d_in[2];
    const float* w_ks  = (const float*)d_in[3];
    const float* w_vs  = (const float*)d_in[4];
    const float* ln1_a = (const float*)d_in[5];
    const float* ln1_b = (const float*)d_in[6];
    const float* w1    = (const float*)d_in[7];
    const float* b1    = (const float*)d_in[8];
    const float* w2    = (const float*)d_in[9];
    const float* b2    = (const float*)d_in[10];
    const float* ln2_a = (const float*)d_in[11];
    const float* ln2_b = (const float*)d_in[12];

    char* ws = (char*)d_ws;
    const size_t MB = 1024 * 1024;
    float*    xf    = (float*)   (ws);              //  8 MB fp32 activations
    float*    tmpO  = (float*)   (ws +  8 * MB);    //  8 MB attn out / ffn out
    _Float16* xh    = (_Float16*)(ws + 16 * MB);    //  4 MB f16 activations
    _Float16* qh    = (_Float16*)(ws + 20 * MB);    //  4 MB
    _Float16* kh    = (_Float16*)(ws + 24 * MB);    //  4 MB
    _Float16* vh    = (_Float16*)(ws + 28 * MB);    //  4 MB (fragment order)
    _Float16* fh    = (_Float16*)(ws + 32 * MB);    //  8 MB f16 FFN hidden
    _Float16* wq16  = (_Float16*)(ws + 40 * MB);    // 512 KB (fragment order)
    _Float16* wk16  = (_Float16*)(ws + 40 * MB + 512 * 1024);
    _Float16* wv16  = (_Float16*)(ws + 41 * MB);
    _Float16* w1t   = (_Float16*)(ws + 42 * MB);    //  1 MB  B[512 x 1024] frag
    _Float16* w2t   = (_Float16*)(ws + 43 * MB);    //  1 MB  B[1024 x 512] frag

    add_pos_kernel<<<(NROWS * D_) / 256, 256, 0, stream>>>(x, pe, xf, xh);

    for (int l = 0; l < 2; ++l) {
        const size_t wqkv_off = (size_t)l * H_ * D_ * DH_;     // 262144
        const size_t wff_off  = (size_t)l * DI_ * D_;          // 524288

        repack_qkv_frag_kernel<<<1024, 256, 0, stream>>>(w_qs + wqkv_off, wq16);
        repack_qkv_frag_kernel<<<1024, 256, 0, stream>>>(w_ks + wqkv_off, wk16);
        repack_qkv_frag_kernel<<<1024, 256, 0, stream>>>(w_vs + wqkv_off, wv16);
        repack_T_frag_kernel<<<2048, 256, 0, stream>>>(w1 + wff_off, w1t, DI_, D_);
        repack_T_frag_kernel<<<2048, 256, 0, stream>>>(w2 + wff_off, w2t, D_, DI_);

        // Q/K projections row-major f16; V projection in B-fragment order
        gemm_wmma_kernel<OUT_F16ROW, false, false>
            <<<dim3(D_ / 128, NROWS / 128), 256, 0, stream>>>(
                xh, wq16, qh, nullptr, NROWS, D_, D_);
        gemm_wmma_kernel<OUT_F16ROW, false, false>
            <<<dim3(D_ / 128, NROWS / 128), 256, 0, stream>>>(
                xh, wk16, kh, nullptr, NROWS, D_, D_);
        gemm_wmma_kernel<OUT_F16FRAG, false, false>
            <<<dim3(D_ / 128, NROWS / 128), 256, 0, stream>>>(
                xh, wv16, vh, nullptr, NROWS, D_, D_);

        // flash attention -> tmpO (fp32, heads concatenated)
        flash_attn_kernel<<<H_ * B_ * (L_ / 16), 32, 0, stream>>>(qh, kh, vh, tmpO);

        // x = LN(attn + residual)
        layernorm_kernel<true><<<NROWS / 8, 256, 0, stream>>>(
            tmpO, xf, ln1_a + l * D_, ln1_b + l * D_, xf, xh);

        // FFN: h = relu(x W1^T + b1) (f16); o = h W2^T + b2 (fp32 -> tmpO)
        gemm_wmma_kernel<OUT_F16ROW, true, true>
            <<<dim3(DI_ / 128, NROWS / 128), 256, 0, stream>>>(
                xh, w1t, fh, b1 + l * DI_, NROWS, DI_, D_);
        gemm_wmma_kernel<OUT_F32, true, false>
            <<<dim3(D_ / 128, NROWS / 128), 256, 0, stream>>>(
                fh, w2t, tmpO, b2 + l * D_, NROWS, D_, DI_);

        // x = LN(o + x); final layer writes straight to d_out
        if (l == 1) {
            layernorm_kernel<false><<<NROWS / 8, 256, 0, stream>>>(
                tmpO, xf, ln2_a + l * D_, ln2_b + l * D_, (float*)d_out, nullptr);
        } else {
            layernorm_kernel<true><<<NROWS / 8, 256, 0, stream>>>(
                tmpO, xf, ln2_a + l * D_, ln2_b + l * D_, xf, xh);
        }
    }
}
